// TAttention_22814866276917
// MI455X (gfx1250) — matrix operand order repaired and checked by reading, not compiled, not run on hardware
//
#include <hip/hip_runtime.h>
#include <hip/hip_bf16.h>

// ---------------------------------------------------------------------------
// MHA for B=4, N=2048, HID=1024, NH=16, HD=64 on gfx1250 (MI455X).
// bf16 WMMA (v_wmma_f32_16x16x32_bf16), fp32 accumulate.
// GEMM weight tiles staged into LDS via global_load_async_to_lds_b128
// (ASYNCcnt path), double-buffered with one barrier per phase.
// ---------------------------------------------------------------------------

typedef __bf16 bf16;
typedef __attribute__((ext_vector_type(16))) __bf16 v16bf;
typedef __attribute__((ext_vector_type(8)))  __bf16 v8bf;
typedef __attribute__((ext_vector_type(8)))  float  v8f;
typedef __attribute__((address_space(3))) bf16 lds_bf16;

#define BATCH 4
#define SEQ   2048
#define HID   1024
#define NHEAD 16
#define HD    64
#define MROWS (BATCH * SEQ)   // 8192

// ---- WMMA fragment loaders -------------------------------------------------
// A-matrix 16x32 bf16 (ISA 7.12.2): lane L holds row M=L%16.
//   lanes 0-15 : K = {0..7, 16..23};  lanes 16-31 : K = {8..15, 24..31}
__device__ __forceinline__ v16bf load_a_frag(const bf16* base, int ld,
                                             int row0, int k0, int lane) {
  int m    = lane & 15;
  int koff = (lane & 16) ? 8 : 0;
  const bf16* p = base + (size_t)(row0 + m) * ld + k0 + koff;
  union { v16bf v; v8bf h[2]; } u;
  u.h[0] = *(const v8bf*)(p);        // K = k0+koff .. +7
  u.h[1] = *(const v8bf*)(p + 16);   // K = k0+16+koff .. +7
  return u.v;
}

// B-matrix 32x16 bf16: lane L holds column N=L%16.
//   lanes 0-15 : K = {0..7, 16..23};  lanes 16-31 : K = {8..15, 24..31}
// `base` holds B column-contiguous: base[col*ld + k] = B[k][col].
__device__ __forceinline__ v16bf load_b_frag(const bf16* base, int ld,
                                             int col0, int k0, int lane) {
  int n     = lane & 15;
  int khalf = (lane & 16) ? 8 : 0;
  const bf16* p = base + (size_t)(col0 + n) * ld + k0 + khalf;
  union { v16bf v; v8bf h[2]; } u;
  u.h[0] = *(const v8bf*)(p);
  u.h[1] = *(const v8bf*)(p + 16);
  return u.v;
}

__device__ __forceinline__ v8f wmma_bf16(v16bf a, v16bf b, v8f c) {
  return __builtin_amdgcn_wmma_f32_16x16x32_bf16(
      /*neg_a=*/false, a, /*neg_b=*/false, b,
      /*c_mod=*/(short)0, c, /*reuse_a=*/false, /*reuse_b=*/false);
}

// ---- async global -> LDS helpers (gfx1250 ASYNCcnt path) -------------------
__device__ __forceinline__ void async_copy_b128(bf16* lds_dst, const bf16* gsrc) {
  unsigned laddr = (unsigned)(unsigned long long)(lds_bf16*)lds_dst;
  unsigned long long gaddr = (unsigned long long)gsrc;
  asm volatile("global_load_async_to_lds_b128 %0, %1, off"
               :: "v"(laddr), "v"(gaddr) : "memory");
}

__device__ __forceinline__ void wait_async0() {
  asm volatile("s_wait_asynccnt 0x0" ::: "memory");
}

// ---- elementwise conversions ----------------------------------------------
__global__ __launch_bounds__(256) void conv_x_kernel(const float* __restrict__ x,
                                                     bf16* __restrict__ xb) {
  size_t i = (size_t)blockIdx.x * 256 + threadIdx.x;   // over 8192*1024
  xb[i] = (bf16)x[i];
}

// W[k][n] (1024x1024 fp32) -> Wt[n][k] bf16 (so B-frag loads are contiguous)
__global__ __launch_bounds__(256) void transpose_w_kernel(const float* __restrict__ W,
                                                          bf16* __restrict__ Wt) {
  int i = blockIdx.x * 256 + threadIdx.x;              // over 1024*1024
  int n = i >> 10;
  int k = i & 1023;
  Wt[i] = (bf16)W[(size_t)k * HID + n];
}

// ---- GEMM: C[8192,1024] = A(bf16,row-major,lda=1024) x W (via Wt[n][k]) ----
// Weight tile (128 cols x 32 k = 8 KB) double-buffered in LDS via async copy.
// mode 0: store bf16 as [b, h, n, d]  (Q, K per-head row-major)
// mode 1: store bf16 as [b, h, d, n]  (V transposed per head)
// mode 2: store fp32 row-major [8192, 1024] (final projection)
__global__ __launch_bounds__(256) void gemm_kernel(const bf16* __restrict__ A,
                                                   const bf16* __restrict__ Wt,
                                                   void* __restrict__ C, int mode) {
  const int tid  = threadIdx.x;
  const int lane = tid & 31;
  const int wave = tid >> 5;
  const int brow0 = blockIdx.x * 128;
  const int bcol0 = blockIdx.y * 128;
  const int row0 = brow0 + (wave & 3) * 32;  // 4 row-groups of 32
  const int wcol = (wave >> 2) * 64;         // 2 col-groups of 64 (within tile)

  // double-buffered weight tile: [2][128 cols][32 k] bf16 (row-major, ld=32)
  __shared__ __align__(16) bf16 btile0[128 * 32];
  __shared__ __align__(16) bf16 btile1[128 * 32];

  // each thread copies exactly two 16B chunks of the 8 KB tile: chunks tid, tid+256
  const int r0 = tid >> 2;             // 0..63
  const int p0 = tid & 3;
  const int r1 = (tid + 256) >> 2;     // 64..127
  const int p1 = p0;
  const bf16* wsrc0 = Wt + (size_t)(bcol0 + r0) * HID + p0 * 8;
  const bf16* wsrc1 = Wt + (size_t)(bcol0 + r1) * HID + p1 * 8;
  bf16* ldst0a = &btile0[r0 * 32 + p0 * 8];
  bf16* ldst0b = &btile0[r1 * 32 + p1 * 8];
  bf16* ldst1a = &btile1[r0 * 32 + p0 * 8];
  bf16* ldst1b = &btile1[r1 * 32 + p1 * 8];

  v8f acc[2][4];
#pragma unroll
  for (int i = 0; i < 2; ++i)
#pragma unroll
    for (int j = 0; j < 4; ++j) acc[i][j] = (v8f)0.0f;

  auto compute = [&](const bf16* bt, int k0) {
    v16bf a0 = load_a_frag(A, HID, row0,      k0, lane);
    v16bf a1 = load_a_frag(A, HID, row0 + 16, k0, lane);
    v16bf b0 = load_b_frag(bt, 32, wcol,      0, lane);
    v16bf b1 = load_b_frag(bt, 32, wcol + 16, 0, lane);
    v16bf b2 = load_b_frag(bt, 32, wcol + 32, 0, lane);
    v16bf b3 = load_b_frag(bt, 32, wcol + 48, 0, lane);
    acc[0][0] = wmma_bf16(a0, b0, acc[0][0]);
    acc[0][1] = wmma_bf16(a0, b1, acc[0][1]);
    acc[0][2] = wmma_bf16(a0, b2, acc[0][2]);
    acc[0][3] = wmma_bf16(a0, b3, acc[0][3]);
    acc[1][0] = wmma_bf16(a1, b0, acc[1][0]);
    acc[1][1] = wmma_bf16(a1, b1, acc[1][1]);
    acc[1][2] = wmma_bf16(a1, b2, acc[1][2]);
    acc[1][3] = wmma_bf16(a1, b3, acc[1][3]);
  };

  // prologue: fill buffer 0 with the k=0 tile
  async_copy_b128(ldst0a, wsrc0);
  async_copy_b128(ldst0b, wsrc1);

#pragma unroll 1
  for (int k0 = 0; k0 < HID; k0 += 64) {
    // ---- phase A: consume btile0 (tile k0), prefetch k0+32 into btile1 ----
    wait_async0();       // this wave's async writes into btile0 landed
    __syncthreads();     // all waves landed + previous-phase ds reads drained
    async_copy_b128(ldst1a, wsrc0 + k0 + 32);
    async_copy_b128(ldst1b, wsrc1 + k0 + 32);
    __builtin_prefetch(A + (size_t)(row0 + (lane & 15)) * HID + k0 + 64, 0, 3);
    compute(btile0, k0);

    // ---- phase B: consume btile1 (tile k0+32), prefetch k0+64 into btile0 -
    wait_async0();
    __syncthreads();
    if (k0 + 64 < HID) {
      async_copy_b128(ldst0a, wsrc0 + k0 + 64);
      async_copy_b128(ldst0b, wsrc1 + k0 + 64);
    }
    compute(btile1, k0 + 32);
  }

  // C/D layout: lane holds col = n0 + (lane&15); rows (lane<16 ? 0..7 : 8..15)
  int rsel = (lane & 16) ? 8 : 0;
  int csel = lane & 15;
#pragma unroll
  for (int i = 0; i < 2; ++i) {
#pragma unroll
    for (int j = 0; j < 4; ++j) {
      int gc = bcol0 + wcol + j * 16 + csel;
#pragma unroll
      for (int v = 0; v < 8; ++v) {
        int gr = row0 + i * 16 + rsel + v;
        float val = acc[i][j][v];
        if (mode == 2) {
          ((float*)C)[(size_t)gr * HID + gc] = val;
        } else {
          int b = gr >> 11, n = gr & 2047;   // gr = b*2048 + n
          int h = gc >> 6,  d = gc & 63;     // gc = h*64 + d
          if (mode == 0)
            ((bf16*)C)[((size_t)((b * NHEAD + h) * SEQ + n) << 6) + d] = (bf16)val;
          else
            ((bf16*)C)[((size_t)((b * NHEAD + h) << 6) + d) * SEQ + n] = (bf16)val;
        }
      }
    }
  }
}

// ---- flash attention: per wave = 16 queries of one (b, h) ------------------
// Q,K: bf16 [b,h,n,d]; Vt: bf16 [b,h,d,n]; O: bf16 [b, n, h*64+d]
__global__ __launch_bounds__(128) void attn_kernel(const bf16* __restrict__ Q,
                                                   const bf16* __restrict__ K,
                                                   const bf16* __restrict__ Vt,
                                                   bf16* __restrict__ O) {
  int lane = threadIdx.x & 31;
  int wave = threadIdx.x >> 5;
  int b = blockIdx.z, h = blockIdx.y;
  int q0 = (blockIdx.x * 4 + wave) * 16;

  const bf16* Qh = Q  + (size_t)(b * NHEAD + h) * SEQ * HD;
  const bf16* Kh = K  + (size_t)(b * NHEAD + h) * SEQ * HD;
  const bf16* Vh = Vt + (size_t)(b * NHEAD + h) * HD * SEQ;

  __shared__ __align__(16) bf16 plds[4][16 * 32];
  bf16* pl = plds[wave];

  // Q fragments (A-matrix), HD=64 -> two K-slices
  v16bf aq0 = load_a_frag(Qh, HD, q0, 0,  lane);
  v16bf aq1 = load_a_frag(Qh, HD, q0, 32, lane);

  float mrow[8], lrow[8];
  v8f o[4];
#pragma unroll
  for (int v = 0; v < 8; ++v) { mrow[v] = -3.0e38f; lrow[v] = 0.0f; }
#pragma unroll
  for (int j = 0; j < 4; ++j) o[j] = (v8f)0.0f;

  const float scale = 0.03125f;  // 1/sqrt(QKV) = 1/32 (reference scales by full dim)
  int rsel = (lane & 16) ? 8 : 0;
  int csel = lane & 15;

#pragma unroll 1
  for (int j0 = 0; j0 < SEQ; j0 += 32) {
    // S = Q * K^T for 32 keys: two 16x16 tiles, each accumulated over HD=64
    v8f s0 = (v8f)0.0f, s1 = (v8f)0.0f;
    v16bf bk;
    bk = load_b_frag(Kh, HD, j0,      0,  lane); s0 = wmma_bf16(aq0, bk, s0);
    bk = load_b_frag(Kh, HD, j0,      32, lane); s0 = wmma_bf16(aq1, bk, s0);
    bk = load_b_frag(Kh, HD, j0 + 16, 0,  lane); s1 = wmma_bf16(aq0, bk, s1);
    bk = load_b_frag(Kh, HD, j0 + 16, 32, lane); s1 = wmma_bf16(aq1, bk, s1);

    // online softmax (row stats across 16 lanes of each half-wave)
    float alpha[8];
#pragma unroll
    for (int v = 0; v < 8; ++v) {
      float a  = s0[v] * scale;
      float c2 = s1[v] * scale;
      float t = fmaxf(a, c2);
      t = fmaxf(t, __shfl_xor(t, 1, 32));
      t = fmaxf(t, __shfl_xor(t, 2, 32));
      t = fmaxf(t, __shfl_xor(t, 4, 32));
      t = fmaxf(t, __shfl_xor(t, 8, 32));
      float mn = fmaxf(mrow[v], t);
      alpha[v] = __expf(mrow[v] - mn);
      mrow[v] = mn;
      float p0 = __expf(a - mn);
      float p1 = __expf(c2 - mn);
      s0[v] = p0; s1[v] = p1;
      float rs = p0 + p1;
      rs += __shfl_xor(rs, 1, 32);
      rs += __shfl_xor(rs, 2, 32);
      rs += __shfl_xor(rs, 4, 32);
      rs += __shfl_xor(rs, 8, 32);
      lrow[v] = lrow[v] * alpha[v] + rs;
    }
#pragma unroll
    for (int j = 0; j < 4; ++j)
#pragma unroll
      for (int v = 0; v < 8; ++v) o[j][v] *= alpha[v];

    // stage P (16x32) through LDS: C-layout -> row-major -> A-layout
#pragma unroll
    for (int v = 0; v < 8; ++v) {
      int m = rsel + v;
      pl[m * 32 + csel]      = (bf16)s0[v];
      pl[m * 32 + 16 + csel] = (bf16)s1[v];
    }
    v16bf ap = load_a_frag(pl, 32, 0, 0, lane);

    // O += P * V (V transposed: B-frag columns are d, K-dim is key index)
#pragma unroll
    for (int jd = 0; jd < 4; ++jd) {
      v16bf bv = load_b_frag(Vh, SEQ, jd * 16, j0, lane);
      o[jd] = wmma_bf16(ap, bv, o[jd]);
    }
  }

  // epilogue: normalize and store concat-head layout [b, n, h*64+d]
#pragma unroll
  for (int v = 0; v < 8; ++v) {
    float inv = 1.0f / lrow[v];
    size_t row = (size_t)(b * SEQ + q0 + rsel + v);
#pragma unroll
    for (int jd = 0; jd < 4; ++jd) {
      O[row * HID + h * HD + jd * 16 + csel] = (bf16)(o[jd][v] * inv);
    }
  }
}

// ---------------------------------------------------------------------------
extern "C" void kernel_launch(void* const* d_in, const int* in_sizes, int n_in,
                              void* d_out, int out_size, void* d_ws, size_t ws_size,
                              hipStream_t stream) {
  const float* x  = (const float*)d_in[0];
  const float* Wq = (const float*)d_in[1];
  const float* Wk = (const float*)d_in[2];
  const float* Wv = (const float*)d_in[3];
  const float* Wo = (const float*)d_in[4];

  char* ws = (char*)d_ws;
  const size_t SZ_XB = (size_t)MROWS * HID * 2;       // 16 MB
  const size_t SZ_W  = (size_t)HID * HID * 2;         // 2 MB each
  const size_t SZ_H  = (size_t)BATCH * NHEAD * SEQ * HD * 2; // 16 MB each

  bf16* xb  = (bf16*)(ws);
  bf16* Wqt = (bf16*)(ws + SZ_XB);
  bf16* Wkt = (bf16*)(ws + SZ_XB + SZ_W);
  bf16* Wvt = (bf16*)(ws + SZ_XB + 2 * SZ_W);
  bf16* Wot = (bf16*)(ws + SZ_XB + 3 * SZ_W);
  bf16* Qb  = (bf16*)(ws + SZ_XB + 4 * SZ_W);
  bf16* Kb  = (bf16*)(ws + SZ_XB + 4 * SZ_W + SZ_H);
  bf16* Vb  = (bf16*)(ws + SZ_XB + 4 * SZ_W + 2 * SZ_H);
  bf16* Ob  = (bf16*)(ws + SZ_XB + 4 * SZ_W + 3 * SZ_H);

  // 1) convert activations + transpose weights to bf16
  conv_x_kernel<<<(MROWS * HID) / 256, 256, 0, stream>>>(x, xb);
  transpose_w_kernel<<<(HID * HID) / 256, 256, 0, stream>>>(Wq, Wqt);
  transpose_w_kernel<<<(HID * HID) / 256, 256, 0, stream>>>(Wk, Wkt);
  transpose_w_kernel<<<(HID * HID) / 256, 256, 0, stream>>>(Wv, Wvt);
  transpose_w_kernel<<<(HID * HID) / 256, 256, 0, stream>>>(Wo, Wot);

  // 2) QKV projections (V stored transposed per head)
  dim3 ggrid(MROWS / 128, HID / 128);
  gemm_kernel<<<ggrid, 256, 0, stream>>>(xb, Wqt, Qb, 0);
  gemm_kernel<<<ggrid, 256, 0, stream>>>(xb, Wkt, Kb, 0);
  gemm_kernel<<<ggrid, 256, 0, stream>>>(xb, Wvt, Vb, 1);

  // 3) flash attention: one wave per 16-query tile
  attn_kernel<<<dim3(SEQ / 64, NHEAD, BATCH), 128, 0, stream>>>(Qb, Kb, Vb, Ob);

  // 4) output projection -> fp32 d_out
  gemm_kernel<<<ggrid, 256, 0, stream>>>(Ob, Wot, (float*)d_out, 2);
}
